// CapsLayer_43525198577821
// MI455X (gfx1250) — matrix operand order repaired and checked by reading, not compiled, not run on hardware
//
#include <hip/hip_runtime.h>

// CapsLayer for MI455X (gfx1250, wave32). Two kernels:
//  K1: u = einsum('bci,cio->bco'), W read exactly once, u stored f16 in
//      K8-grouped layout u8[b][c>>3][o][c&7]  (64 MB in d_ws).
//  K2 (per batch): C[b]=u[b]^T u[b] via v_wmma_f32_16x16x32_f16, upper
//      triangle only, async double-buffered LDS chunks; then power iteration
//      for the top eigenvector (C is PSD Gram), sign-canonicalized.

typedef float    v8f  __attribute__((ext_vector_type(8)));
typedef _Float16 v8h  __attribute__((ext_vector_type(8)));
typedef _Float16 v16h __attribute__((ext_vector_type(16)));

#define BATCH   64
#define CAPS    2048
#define IDIM    16
#define DOUT    256
#define CHUNK   32                  // capsules per K2 chunk (= one K step of 32)
#define NCHUNK  (CAPS / CHUNK)      // 64
#define NKG     (CAPS / 8)          // 256 K-groups of 8 capsules
#define CST     260                 // C LDS row stride (float4-aligned, near-conflict-free)
#define NTILE   17                  // upper-tri 16x16 tiles per wave (136 / 8 waves)
#define PITERS  64

#define U8_PER_B   ((size_t)NKG * DOUT * 8)          // 524288 halves per batch
#define CHUNK_H    (CHUNK * DOUT)                    // 8192 halves per chunk slab
#define LDS2_C     (256 * CST)                       // 66560 floats
#define LDS2_TOT   (LDS2_C + 256 + 64)

// ---------------- Kernel 1: u (VALU GEMV batch), W read once ----------------
__global__ __launch_bounds__(256)
void u_kernel(const float* __restrict__ x,      // [B][CAPS][IDIM]
              const float* __restrict__ w,      // [CAPS][IDIM][DOUT]
              _Float16*    __restrict__ u8)     // [B][NKG][DOUT][8]
{
    __shared__ float    Xl[8 * 64 * 16];        // [c][b][i]  32 KB
    __shared__ _Float16 Rl[64 * 256 * 8];       // [b][o][c]  256 KB

    const int g   = blockIdx.x;                 // capsule octet: caps 8g..8g+7
    const int tid = threadIdx.x;

    // stage X for the 8 capsules, all 64 batches (coalesced float4)
#pragma unroll
    for (int r = 0; r < 8; ++r) {
        const int pid = tid + 256 * r;
        const int q = pid & 3, c = (pid >> 2) & 7, b = pid >> 5;
        const float4 vx = *(const float4*)(x + (size_t)b * (CAPS * IDIM)
                                             + (size_t)(g * 8 + c) * IDIM + q * 4);
        *(float4*)(&Xl[((c << 6) + b) * 16 + q * 4]) = vx;
    }
    __syncthreads();

    const int o = tid;
    for (int c = 0; c < 8; ++c) {
        float wv[16];
#pragma unroll
        for (int i = 0; i < 16; ++i)
            wv[i] = w[((size_t)(g * 8 + c) * IDIM + i) * DOUT + o];
        for (int b = 0; b < 64; ++b) {
            float s = 0.f;
#pragma unroll
            for (int i = 0; i < 16; ++i) s += wv[i] * Xl[((c << 6) + b) * 16 + i];
            Rl[((b << 8) + o) * 8 + c] = (_Float16)s;
        }
    }
    __syncthreads();

    // copy out: u8[b][kg=g][o][0..7], fully coalesced b128
    for (int r = 0; r < 64; ++r) {
        const int pid = tid + 256 * r;
        const int b = pid >> 8, oo = pid & 255;
        *(float4*)(u8 + (size_t)b * U8_PER_B + (size_t)g * (DOUT * 8) + oo * 8) =
            *(const float4*)(&Rl[((b << 8) + oo) * 8]);
    }
}

// --------------- Kernel 2: Gram (f16 WMMA) + top eigenvector ----------------
// A frag (16x32 f16): lane half h=lane>>4 -> K octets {h, h+2}, col = ib*16+(lane&15)
// B frag (32x16 f16): lane half h        -> K octets {h, h+2}, col = jb*16+(lane&15)
__device__ __forceinline__ v16h ldfragA(const _Float16* buf, int blk, int lane) {
    const int col = (blk << 4) + (lane & 15);
    const int h   = lane >> 4;
    const v8h lo = *(const v8h*)(buf + (size_t)(h    ) * (DOUT * 8) + col * 8);
    const v8h hi = *(const v8h*)(buf + (size_t)(h + 2) * (DOUT * 8) + col * 8);
    return __builtin_shufflevector(lo, hi, 0,1,2,3,4,5,6,7,8,9,10,11,12,13,14,15);
}
__device__ __forceinline__ v16h ldfragB(const _Float16* buf, int blk, int lane) {
    const int col = (blk << 4) + (lane & 15);
    const int h   = lane >> 4;
    const v8h lo = *(const v8h*)(buf + (size_t)(h    ) * (DOUT * 8) + col * 8);
    const v8h hi = *(const v8h*)(buf + (size_t)(h + 2) * (DOUT * 8) + col * 8);
    return __builtin_shufflevector(lo, hi, 0,1,2,3,4,5,6,7,8,9,10,11,12,13,14,15);
}

__global__ __launch_bounds__(256)
void gram_eig_kernel(const _Float16* __restrict__ u8,   // [B][NKG][DOUT][8]
                     float*          __restrict__ out)  // [B][DOUT]
{
    __shared__ float lds[LDS2_TOT];
    float*     Clds = lds;                       // [256][CST] (valid after accumulation)
    _Float16*  bufs = (_Float16*)lds;            // 2 x CHUNK_H halves alias the C region
    float*     vld  = lds + LDS2_C;
    float*     red  = vld + 256;

    const int b    = blockIdx.x;
    const int tid  = threadIdx.x;
    const int lane = tid & 31;
    const int wave = __builtin_amdgcn_readfirstlane(tid >> 5);

    // decode this wave's 17 upper-triangular tile coordinates (scalar)
    int ti[NTILE], tj[NTILE];
    {
        int t = wave * NTILE;
        int i = 0;
        while (t >= 16 - i) { t -= 16 - i; ++i; }
        int j = i + t;
        for (int k = 0; k < NTILE; ++k) {
            ti[k] = i; tj[k] = j;
            if (++j > 15) { ++i; j = i; }
        }
    }

    v8f acc[NTILE];
#pragma unroll
    for (int k = 0; k < NTILE; ++k)
#pragma unroll
        for (int r = 0; r < 8; ++r) acc[k][r] = 0.0f;

    const _Float16* ub = u8 + (size_t)b * U8_PER_B;

    // async-copy one 16KB chunk slab (contiguous) into LDS buffer `parity`
    auto issue_chunk = [&](int ch) {
        const int parity = ch & 1;
        const unsigned ldsbase =
            (unsigned)(uintptr_t)(bufs + (size_t)parity * CHUNK_H);
        const unsigned long long gbase =
            (unsigned long long)(uintptr_t)(ub + (size_t)ch * CHUNK_H);
#pragma unroll
        for (int r = 0; r < 4; ++r) {
            const int p = tid + 256 * r;
            const unsigned           loff = ldsbase + (unsigned)p * 16u;
            const unsigned long long ga   = gbase + (unsigned long long)p * 16ull;
            asm volatile("global_load_async_to_lds_b128 %0, %1, off"
                         :: "v"(loff), "v"(ga) : "memory");
        }
    };
    auto gram_step = [&](const _Float16* buf) {
#pragma unroll
        for (int k = 0; k < NTILE; ++k) {
            const v16h a  = ldfragA(buf, ti[k], lane);
            const v16h bf = ldfragB(buf, tj[k], lane);
            acc[k] = __builtin_amdgcn_wmma_f32_16x16x32_f16(
                false, a, false, bf, (short)0, acc[k], false, false);
        }
    };

    issue_chunk(0);
#pragma unroll 1
    for (int ch = 0; ch < NCHUNK - 1; ++ch) {
        issue_chunk(ch + 1);                               // overlaps this chunk's WMMA
        asm volatile("s_wait_asynccnt 0x4" ::: "memory");  // chunk ch landed (in-order)
        __syncthreads();
        gram_step(bufs + (size_t)(ch & 1) * CHUNK_H);
        __syncthreads();                                   // frag reads done before overwrite
    }
    asm volatile("s_wait_asynccnt 0x0" ::: "memory");
    __syncthreads();
    gram_step(bufs + (size_t)((NCHUNK - 1) & 1) * CHUNK_H);
    __syncthreads();                                       // buffers dead; LDS becomes C

    // spill C tiles to LDS (mirror lower triangle)
    {
        const int n     = lane & 15;
        const int mhalf = (lane >> 4) << 3;
#pragma unroll
        for (int k = 0; k < NTILE; ++k) {
            const int ib = ti[k] << 4, jb = tj[k] << 4;
#pragma unroll
            for (int r = 0; r < 8; ++r) {
                const int m = r + mhalf;
                const float val = acc[k][r];
                Clds[(ib + m) * CST + (jb + n)] = val;
                if (ib != jb)
                    Clds[(jb + n) * CST + (ib + m)] = val;
            }
        }
    }

    // power iteration (float4 matvec, stride 260 keeps banks happy)
    vld[tid] = 1.0f;
    __syncthreads();

    for (int it = 0; it < PITERS; ++it) {
        float s = 0.0f;
        const float* crow = Clds + tid * CST;
        for (int e = 0; e < 64; ++e) {
            const float4 cv = *(const float4*)(crow + 4 * e);
            const float4 vv = *(const float4*)(vld + 4 * e);
            s += cv.x * vv.x + cv.y * vv.y + cv.z * vv.z + cv.w * vv.w;
        }
        float q = s * s;
#pragma unroll
        for (int off = 16; off > 0; off >>= 1) q += __shfl_xor(q, off);
        if (lane == 0) red[wave] = q;
        __syncthreads();
        if (tid == 0) {
            float tot = 0.0f;
            for (int wv = 0; wv < 8; ++wv) tot += red[wv];
            red[8] = rsqrtf(tot);
        }
        __syncthreads();
        vld[tid] = s * red[8];
        __syncthreads();
    }

    // sign canonicalization: first argmax of |v| flips positive
    if (tid == 0) {
        int best = 0; float bv = fabsf(vld[0]);
        for (int e = 1; e < 256; ++e) {
            const float a = fabsf(vld[e]);
            if (a > bv) { bv = a; best = e; }
        }
        const float sv = vld[best];
        red[9] = (sv > 0.0f) ? 1.0f : ((sv < 0.0f) ? -1.0f : 0.0f);
    }
    __syncthreads();
    out[(size_t)b * DOUT + tid] = vld[tid] * red[9];
}

extern "C" void kernel_launch(void* const* d_in, const int* in_sizes, int n_in,
                              void* d_out, int out_size, void* d_ws, size_t ws_size,
                              hipStream_t stream) {
    (void)in_sizes; (void)n_in; (void)out_size; (void)ws_size;
    const float* caps = (const float*)d_in[0];   // [64][2048][16] fp32
    const float* wts  = (const float*)d_in[1];   // [2048][16][256] fp32
    float*       out  = (float*)d_out;           // [64][256] fp32
    _Float16*    u8   = (_Float16*)d_ws;         // needs 64 MB scratch

    hipLaunchKernelGGL(u_kernel, dim3(NKG), dim3(256), 0, stream, caps, wts, u8);
    hipLaunchKernelGGL(gram_eig_kernel, dim3(BATCH), dim3(256), 0, stream, u8, out);
}
